// LocalBandSimilarityBlock_81801947120109
// MI455X (gfx1250) — matrix operand order repaired and checked by reading, not compiled, not run on hardware
//
#include <hip/hip_runtime.h>
#include <math.h>

// ---------------------------------------------------------------------------
// LocalBandSimilarityBlock for MI455X (gfx1250, wave32, WMMA, async-LDS).
// N=4096 tokens (64x64 grid), D=768, radius 2 => <=24 neighbors per token.
// GEMMs (58 GFLOP): v_wmma_f32_16x16x32_f16, f16 operands, double-buffered
// global->LDS async staging (ASYNCcnt). Attention (0.3 GFLOP, sparse):
// wave32 VALU, one wave per token (<=25 candidates -> one per lane).
// ---------------------------------------------------------------------------

typedef __attribute__((ext_vector_type(16))) _Float16 v16h;
typedef __attribute__((ext_vector_type(8)))  _Float16 h8;
typedef __attribute__((ext_vector_type(8)))  float    v8f;
typedef __attribute__((ext_vector_type(4)))  float    f4;

#define N_TOK 4096
#define DIM   768
#define HDIM  3072
#define SIDE  64
#define NEGV  (-1e30f)

// GEMM tiling: block = 256 thr = 8 waves (2 M-rows x 4 N-cols),
// block tile 64(M) x 256(N), K-step 32, wave tile 32x64 = 8 WMMAs/step.
#define BM    64
#define BN    256
#define BK    32
#define LDST  40                    // LDS row stride in halves (80B: pad vs bank conflicts)
#define BOFF  (BM * LDST)           // B tile offset in halves (2560)
#define BUFH  (BOFF + BN * LDST)    // halves per buffer (12800)

// ---- CDNA5 async global->LDS copy (16B), tracked by ASYNCcnt ---------------
__device__ __forceinline__ void async_cp16(unsigned lds_byte_addr, const void* g) {
  asm volatile("global_load_async_to_lds_b128 %0, %1, off"
               :: "v"(lds_byte_addr), "v"((unsigned long long)(uintptr_t)g)
               : "memory");
}
#define WAIT_ASYNC(n) asm volatile("s_wait_asynccnt " #n ::: "memory")

// ---- fragment loaders (CDNA5 16x16x32 f16 layouts, ISA 7.12.2) -------------
// A: lane L -> row m=L&15, kb=(L>=16)?8:0; elems 0..7 = K kb..kb+7,
//    elems 8..15 = K kb+16..kb+23 (both contiguous 16B runs).
__device__ __forceinline__ v16h loadA_frag(const _Float16* p) {
  h8 lo = *(const h8*)p;
  h8 hi = *(const h8*)(p + 16);
  v16h r;
#pragma unroll
  for (int i = 0; i < 8; ++i) { r[i] = lo[i]; r[8 + i] = hi[i]; }
  return r;
}
// B: lane L -> col n=L&15, kb=(L>=16)?8:0; elems 0..7 = K kb..kb+7,
//    elems 8..15 = K kb+16..kb+23.
__device__ __forceinline__ v16h loadB_frag(const _Float16* p) {
  h8 lo = *(const h8*)p;
  h8 hi = *(const h8*)(p + 16);
  v16h r;
#pragma unroll
  for (int i = 0; i < 8; ++i) { r[i] = lo[i]; r[8 + i] = hi[i]; }
  return r;
}

__device__ __forceinline__ float gelu_exact(float x) {
  return 0.5f * x * (1.0f + erff(x * 0.70710678118654752f));
}

// ---------------------------------------------------------------------------
// C[M,N] = act(A[M,K] @ B[N,K]^T + bias[N]) (+ res[M,N]); A,B f16 row-major,
// both K-contiguous. Double-buffered async LDS staging, last K-step peeled
// so the steady-state loop has no conditional waits.
// ---------------------------------------------------------------------------
template <typename OT, bool GELU, bool RES>
__global__ __launch_bounds__(256)
void wmma_gemm_kernel(const _Float16* __restrict__ A, const _Float16* __restrict__ B,
                      const float* __restrict__ bias, const float* __restrict__ res,
                      OT* __restrict__ C, int M, int N, int K) {
  __shared__ __align__(16) _Float16 smem[2 * BUFH];   // 51.2 KB

  const int tid  = threadIdx.x;
  const int lane = tid & 31;
  const int wave = tid >> 5;
  const int wm   = wave >> 2;              // 0..1
  const int wn   = wave & 3;               // 0..3
  const int bm0  = blockIdx.y * BM;
  const int bn0  = blockIdx.x * BN;
  const int m0l  = wm * 32;
  const int n0l  = wn * 64;

  const int kbA  = (lane >> 4) ? 8 : 0;
  const int kbB  = (lane >> 4) ? 8 : 0;
  const int nc   = lane & 15;
  const int mr   = lane & 15;

  const unsigned lds_base = (unsigned)(uintptr_t)smem;

  // per-thread staging assignments (5 x 16B per K-step: 1 A chunk + 4 B chunks)
  const int arow = tid >> 2, ach = tid & 3;              // A: 64 rows x 4 chunks
  const _Float16* gA = A + (size_t)(bm0 + arow) * K + ach * 8;
  const unsigned  lA = (unsigned)((arow * LDST + ach * 8) * 2);

  v8f acc[2][4];
#pragma unroll
  for (int mi = 0; mi < 2; ++mi)
#pragma unroll
    for (int ni = 0; ni < 4; ++ni) acc[mi][ni] = (v8f){};

  const int KT = K / BK;

  // stage K-tile kt into buffer (kt&1)
  auto stage = [&](int kt) {
    const int k0  = kt * BK;
    const unsigned bufb = (unsigned)((kt & 1) * BUFH * 2);
    async_cp16(lds_base + bufb + lA, gA + k0);
#pragma unroll
    for (int t = 0; t < 4; ++t) {
      const int c = t * 256 + tid;
      const int brow = c >> 2, bch = c & 3;              // B: 256 rows x 4 chunks
      async_cp16(lds_base + bufb + (unsigned)((BOFF + brow * LDST + bch * 8) * 2),
                 B + (size_t)(bn0 + brow) * K + k0 + bch * 8);
    }
  };

  // one K-step of WMMAs out of buffer (kt&1)
  auto compute = [&](int kt) {
    const _Float16* sb = smem + (kt & 1) * BUFH;
    v16h afr[2];
#pragma unroll
    for (int mi = 0; mi < 2; ++mi)
      afr[mi] = loadA_frag(sb + (m0l + mi * 16 + mr) * LDST + kbA);
#pragma unroll
    for (int ni = 0; ni < 4; ++ni) {
      v16h bf = loadB_frag(sb + BOFF + (n0l + ni * 16 + nc) * LDST + kbB);
      acc[0][ni] = __builtin_amdgcn_wmma_f32_16x16x32_f16(false, afr[0], false, bf,
                                                          (short)0, acc[0][ni], false, false);
      acc[1][ni] = __builtin_amdgcn_wmma_f32_16x16x32_f16(false, afr[1], false, bf,
                                                          (short)0, acc[1][ni], false, false);
    }
  };

  stage(0);

  // steady state: unconditionally stage next tile, retire current, compute.
  for (int kt = 0; kt < KT - 1; ++kt) {
    stage(kt + 1);
    WAIT_ASYNC(5);          // current tile done; next tile's 5 remain in flight
    __syncthreads();        // tile (kt) visible to all waves
    compute(kt);
    __syncthreads();        // compute done before this buffer is restaged
  }
  // peeled last K-step
  WAIT_ASYNC(0);
  __syncthreads();
  compute(KT - 1);

  const int moff = (lane >> 4) ? 8 : 0;
#pragma unroll
  for (int mi = 0; mi < 2; ++mi)
#pragma unroll
    for (int ni = 0; ni < 4; ++ni)
#pragma unroll
      for (int r = 0; r < 8; ++r) {
        const int m = bm0 + m0l + mi * 16 + r + moff;
        const int n = bn0 + n0l + ni * 16 + nc;
        float v = acc[mi][ni][r] + bias[n];
        if (RES)  v += res[(size_t)m * N + n];
        if (GELU) v = gelu_exact(v);
        C[(size_t)m * N + n] = (OT)v;
      }
}

// ---------------------------------------------------------------------------
// fp32 -> f16 weight conversion (vectorized, n multiple of 2048 per launch)
// ---------------------------------------------------------------------------
__global__ __launch_bounds__(256)
void cvt_f16_kernel(const float* __restrict__ s, _Float16* __restrict__ d) {
  const int i = (blockIdx.x * 256 + threadIdx.x) * 8;
  f4 a = *(const f4*)(s + i);
  f4 b = *(const f4*)(s + i + 4);
  h8 o;
#pragma unroll
  for (int j = 0; j < 4; ++j) { o[j] = (_Float16)a[j]; o[4 + j] = (_Float16)b[j]; }
  *(h8*)(d + i) = o;
}

// ---------------------------------------------------------------------------
// LayerNorm over D=768, one block (256 thr = 8 waves) per row, f16 out.
// ---------------------------------------------------------------------------
__global__ __launch_bounds__(256)
void ln_kernel(const float* __restrict__ x, const float* __restrict__ g,
               const float* __restrict__ b, _Float16* __restrict__ out) {
  const int row  = blockIdx.x;
  const int tid  = threadIdx.x;
  const int lane = tid & 31;
  const int wave = tid >> 5;
  const float* xr = x + (size_t)row * DIM;

  float v0 = xr[tid], v1 = xr[tid + 256], v2 = xr[tid + 512];
  float s  = v0 + v1 + v2;
  float ss = v0 * v0 + v1 * v1 + v2 * v2;
#pragma unroll
  for (int off = 16; off; off >>= 1) {
    s  += __shfl_xor(s,  off, 32);
    ss += __shfl_xor(ss, off, 32);
  }
  __shared__ float red[16];
  if (lane == 0) { red[wave] = s; red[wave + 8] = ss; }
  __syncthreads();
  float S = 0.f, SS = 0.f;
#pragma unroll
  for (int i = 0; i < 8; ++i) { S += red[i]; SS += red[i + 8]; }
  const float mu   = S * (1.0f / DIM);
  const float var  = SS * (1.0f / DIM) - mu * mu;
  const float rstd = rsqrtf(var + 1e-5f);

  _Float16* orow = out + (size_t)row * DIM;
  orow[tid]       = (_Float16)((v0 - mu) * rstd * g[tid]       + b[tid]);
  orow[tid + 256] = (_Float16)((v1 - mu) * rstd * g[tid + 256] + b[tid + 256]);
  orow[tid + 512] = (_Float16)((v2 - mu) * rstd * g[tid + 512] + b[tid + 512]);
}

// ---------------------------------------------------------------------------
// Local-band attention: one wave32 per token. 25 candidate offsets (5x5);
// lane c owns candidate c's score; dots computed cooperatively (24 dims/lane).
// f16 output (feeds the Wo projection GEMM directly).
// ---------------------------------------------------------------------------
__global__ __launch_bounds__(256)
void attn_kernel(const float* __restrict__ q, const float* __restrict__ k,
                 const float* __restrict__ v, const int* __restrict__ grid,
                 _Float16* __restrict__ out) {
  const int lane  = threadIdx.x & 31;
  const int wave  = threadIdx.x >> 5;
  const int token = blockIdx.x * 8 + wave;

  const float scale = 0.03608439182435161f;  // 1/sqrt(768)
  const int gx = grid[2 * token];
  const int gy = grid[2 * token + 1];

  float qreg[24];
  {
    const float* qp = q + (size_t)token * DIM + lane;
#pragma unroll
    for (int j = 0; j < 24; ++j) qreg[j] = qp[j * 32];
  }

  float myscore = NEGV;
  int   mynbr   = -1;

  for (int c = 0; c < 25; ++c) {          // wave-uniform loop & branch
    const int dy = c / 5 - 2, dx = c % 5 - 2;
    const int nx = gx + dx, ny = gy + dy;
    const bool valid = !(dx == 0 && dy == 0) &&
                       nx >= 0 && nx < SIDE && ny >= 0 && ny < SIDE;
    if (valid) {
      const int nbr = ny * SIDE + nx;
      const float* kp = k + (size_t)nbr * DIM + lane;
      float p = 0.f;
#pragma unroll
      for (int j = 0; j < 24; ++j) p = fmaf(qreg[j], kp[j * 32], p);
#pragma unroll
      for (int off = 16; off; off >>= 1) p += __shfl_xor(p, off, 32);
      if (lane == c) { myscore = p * scale; mynbr = nbr; }
    }
  }

  // softmax across lanes (invalid lanes hold NEGV / nbr=-1 -> weight 0)
  float mx = myscore;
#pragma unroll
  for (int off = 16; off; off >>= 1) mx = fmaxf(mx, __shfl_xor(mx, off, 32));
  float e = (mynbr >= 0) ? __expf(myscore - mx) : 0.f;
  float sum = e;
#pragma unroll
  for (int off = 16; off; off >>= 1) sum += __shfl_xor(sum, off, 32);

  float acc[24];
#pragma unroll
  for (int j = 0; j < 24; ++j) acc[j] = 0.f;

  if (sum > 0.f) {
    const float inv = 1.0f / sum;
    for (int c = 0; c < 25; ++c) {
      const float w  = __shfl(e, c, 32) * inv;
      const int   nb = __shfl(mynbr, c, 32);
      if (nb >= 0) {
        const float* vp = v + (size_t)nb * DIM + lane;
#pragma unroll
        for (int j = 0; j < 24; ++j) acc[j] = fmaf(w, vp[j * 32], acc[j]);
      }
    }
  } else {  // no neighbors: pass v through
    const float* vp = v + (size_t)token * DIM + lane;
#pragma unroll
    for (int j = 0; j < 24; ++j) acc[j] = vp[j * 32];
  }

  _Float16* op = out + (size_t)token * DIM + lane;
#pragma unroll
  for (int j = 0; j < 24; ++j) op[j * 32] = (_Float16)acc[j];
}

// ---------------------------------------------------------------------------
extern "C" void kernel_launch(void* const* d_in, const int* in_sizes, int n_in,
                              void* d_out, int out_size, void* d_ws, size_t ws_size,
                              hipStream_t stream) {
  const float* x     = (const float*)d_in[0];
  const int*   grid  = (const int*)  d_in[1];
  const float* Wq    = (const float*)d_in[2];
  const float* bq    = (const float*)d_in[3];
  const float* Wk    = (const float*)d_in[4];
  const float* bk    = (const float*)d_in[5];
  const float* Wv    = (const float*)d_in[6];
  const float* bv    = (const float*)d_in[7];
  const float* Wo    = (const float*)d_in[8];
  const float* bo    = (const float*)d_in[9];
  const float* g1    = (const float*)d_in[10];
  const float* beta1 = (const float*)d_in[11];
  const float* g2    = (const float*)d_in[12];
  const float* beta2 = (const float*)d_in[13];
  const float* W1    = (const float*)d_in[14];
  const float* b1    = (const float*)d_in[15];
  const float* W2    = (const float*)d_in[16];
  const float* b2    = (const float*)d_in[17];
  float* out = (float*)d_out;

  // workspace layout (~108 MB): f16 region first, then f32 region.
  const size_t ND = (size_t)N_TOK * DIM;       // 3,145,728
  const size_t WD = (size_t)DIM * DIM;         //   589,824
  const size_t WH = (size_t)HDIM * DIM;        // 2,359,296
  _Float16* wq16  = (_Float16*)d_ws;
  _Float16* wk16  = wq16 + WD;
  _Float16* wv16  = wk16 + WD;
  _Float16* wo16  = wv16 + WD;
  _Float16* w116  = wo16 + WD;
  _Float16* w216  = w116 + WH;
  _Float16* h16   = w216 + WH;
  _Float16* attn16= h16  + ND;
  _Float16* h216  = attn16 + ND;
  _Float16* mid16 = h216 + ND;                 // [N, 4D]
  float*    q     = (float*)(mid16 + (size_t)N_TOK * HDIM);
  float*    kk    = q  + ND;
  float*    vv    = kk + ND;
  float*    x1    = vv + ND;

  const dim3 blk(256);
  const dim3 gD(DIM  / BN, N_TOK / BM);   // (3, 64)
  const dim3 gH(HDIM / BN, N_TOK / BM);   // (12, 64)

  // 0) weights -> f16 (once per launch; deterministic)
  cvt_f16_kernel<<<WD / 2048, blk, 0, stream>>>(Wq, wq16);
  cvt_f16_kernel<<<WD / 2048, blk, 0, stream>>>(Wk, wk16);
  cvt_f16_kernel<<<WD / 2048, blk, 0, stream>>>(Wv, wv16);
  cvt_f16_kernel<<<WD / 2048, blk, 0, stream>>>(Wo, wo16);
  cvt_f16_kernel<<<WH / 2048, blk, 0, stream>>>(W1, w116);
  cvt_f16_kernel<<<WH / 2048, blk, 0, stream>>>(W2, w216);

  // 1) h = LN(x; g1, beta1)  (f16)
  ln_kernel<<<N_TOK, blk, 0, stream>>>(x, g1, beta1, h16);
  // 2) q,k,v = h @ W^T + b   (f32 out for attention math)
  wmma_gemm_kernel<float, false, false><<<gD, blk, 0, stream>>>(
      h16, wq16, bq, nullptr, q, N_TOK, DIM, DIM);
  wmma_gemm_kernel<float, false, false><<<gD, blk, 0, stream>>>(
      h16, wk16, bk, nullptr, kk, N_TOK, DIM, DIM);
  wmma_gemm_kernel<float, false, false><<<gD, blk, 0, stream>>>(
      h16, wv16, bv, nullptr, vv, N_TOK, DIM, DIM);
  // 3) local masked softmax attention (f16 out)
  attn_kernel<<<N_TOK / 8, blk, 0, stream>>>(q, kk, vv, grid, attn16);
  // 4) x1 = x + attn @ Wo^T + bo
  wmma_gemm_kernel<float, false, true><<<gD, blk, 0, stream>>>(
      attn16, wo16, bo, x, x1, N_TOK, DIM, DIM);
  // 5) h2 = LN(x1; g2, beta2)  (f16)
  ln_kernel<<<N_TOK, blk, 0, stream>>>(x1, g2, beta2, h216);
  // 6) mid = gelu(h2 @ W1^T + b1)  (f16 out)
  wmma_gemm_kernel<_Float16, true, false><<<gH, blk, 0, stream>>>(
      h216, w116, b1, nullptr, mid16, N_TOK, HDIM, DIM);
  // 7) out = x1 + mid @ W2^T + b2
  wmma_gemm_kernel<float, false, true><<<gD, blk, 0, stream>>>(
      mid16, w216, b2, x1, out, N_TOK, DIM, HDIM);
}
